// CrossAttention_7559142441748
// MI455X (gfx1250) — matrix operand order repaired and checked by reading, not compiled, not run on hardware
//
#include <hip/hip_runtime.h>

typedef __attribute__((ext_vector_type(16))) __bf16 v16bf;
typedef __attribute__((ext_vector_type(8)))  float  v8f;

#define Bc 8
#define CC 64
#define KN 16
#define MM 8192
#define MT 16          // points (m values) per block
#define NW 8           // waves per block (256 threads, wave32)
#define PW (MT / NW)   // points per wave
#define MP 17          // padded m-extent in LDS (bank-conflict-free strided reads)

__device__ __forceinline__ void lds_wait() {
  asm volatile("s_wait_dscnt 0" ::: "memory");
}
__device__ __forceinline__ void async_wait() {
  asm volatile("s_wait_asynccnt 0" ::: "memory");
}
__device__ __forceinline__ unsigned lds_offset(const void* p) {
  // generic (flat) pointers to LDS carry the LDS byte offset in their low 32 bits
  return (unsigned)(reinterpret_cast<uintptr_t>(p));
}

__device__ __forceinline__ v8f wmma_bf16(v16bf a, v16bf b, v8f c) {
  return __builtin_amdgcn_wmma_f32_16x16x32_bf16(false, a, false, b, (short)0, c,
                                                 false, false);
}

// A-fragment (16x32 bf16) from an LDS bf16 [16][64] region, K-chunk h.
__device__ __forceinline__ v16bf load_A(const __bf16* s, int h, int lane) {
  int row = lane & 15;
  int bc  = 32 * h + 8 * (lane >> 4);
  const unsigned* p = (const unsigned*)(s + row * 64 + bc);
  union { v16bf v; unsigned u[8]; } r;
#pragma unroll
  for (int j = 0; j < 4; ++j) { r.u[j] = p[j]; r.u[4 + j] = p[8 + j]; }
  return r.v;
}

// A-fragment built from the f32 feature tile in LDS, layout [c*16+k][MP] (m padded).
__device__ __forceinline__ v16bf load_A_feat(const float* sf, int p, int h, int lane) {
  int row = lane & 15;                 // neighbor k
  int bc  = 32 * h + 8 * (lane >> 4);  // channel base
  union { v16bf v; __bf16 u[16]; } r;
#pragma unroll
  for (int e = 0; e < 8; ++e) {
    r.u[e]     = (__bf16)sf[((bc + e) * 16 + row) * MP + p];
    r.u[8 + e] = (__bf16)sf[((bc + 16 + e) * 16 + row) * MP + p];
  }
  return r.v;
}

// B-fragment (32x16 bf16) for W stored [o][c] (row-major, 64x64) in LDS:
// out tile t (o = 16t+n), K-chunk h (c = 32h + 8*lanegroup + e, e<8; +16 for e>=8).
__device__ __forceinline__ v16bf load_Bw(const __bf16* w, int t, int h, int lane) {
  int o  = 16 * t + (lane & 15);
  int c0 = 32 * h + 8 * (lane >> 4);
  const unsigned* p = (const unsigned*)(w + o * 64 + c0);
  union { v16bf v; unsigned u[8]; } r;
#pragma unroll
  for (int j = 0; j < 4; ++j) { r.u[j] = p[j]; r.u[4 + j] = p[8 + j]; }
  return r.v;
}

// ---- small xyz (3->64) projections built directly in fragment layouts ----
__device__ __forceinline__ v16bf make_qA(const float* W, const float* bb,
                                         const float* xyz, int h, int lane) {
  int row = lane & 15, bc = 32 * h + 8 * (lane >> 4);
  float x0 = xyz[row * 4 + 0], x1 = xyz[row * 4 + 1], x2 = xyz[row * 4 + 2];
  union { v16bf v; __bf16 u[16]; } r;
#pragma unroll
  for (int e = 0; e < 8; ++e) {
    int c = bc + e;
    r.u[e]     = (__bf16)(bb[c] + x0 * W[3 * c] + x1 * W[3 * c + 1] + x2 * W[3 * c + 2]);
    c = bc + 16 + e;
    r.u[8 + e] = (__bf16)(bb[c] + x0 * W[3 * c] + x1 * W[3 * c + 1] + x2 * W[3 * c + 2]);
  }
  return r.v;
}

__device__ __forceinline__ v16bf make_kB(const float* W, const float* bb,
                                         const float* xyz, int h, int lane) {
  int j = lane & 15, c0 = 32 * h + 8 * (lane >> 4);
  float x0 = xyz[j * 4 + 0], x1 = xyz[j * 4 + 1], x2 = xyz[j * 4 + 2];
  union { v16bf v; __bf16 u[16]; } r;
#pragma unroll
  for (int e = 0; e < 16; ++e) {
    int c = c0 + (e < 8 ? e : e + 8);
    r.u[e] = (__bf16)(bb[c] + x0 * W[3 * c] + x1 * W[3 * c + 1] + x2 * W[3 * c + 2]);
  }
  return r.v;
}

__device__ __forceinline__ v16bf make_vB(const float* W, const float* bb,
                                         const float* xyz, int t, int lane) {
  int c = 16 * t + (lane & 15);
  int hi = lane >> 4;   // j = 8*lanegroup + e for e<8; elements 8..15 (j=16..31) -> zero padding
  float w0 = W[3 * c], w1 = W[3 * c + 1], w2 = W[3 * c + 2], b0 = bb[c];
  union { v16bf v; __bf16 u[16]; } r;
#pragma unroll
  for (int e = 0; e < 16; ++e) {
    int jj = 8 * hi + (e & 7);
    float val = b0 + xyz[jj * 4 + 0] * w0 + xyz[jj * 4 + 1] * w1 + xyz[jj * 4 + 2] * w2;
    r.u[e] = (e >= 8) ? (__bf16)0.f : (__bf16)val;
  }
  return r.v;
}

// relu(softmax(q k^T / 8) v) for 3-channel xyz inputs; result in D layout (4 tiles).
__device__ __forceinline__ void self_att(const float* qW, const float* qb,
                                         const float* kW, const float* kb,
                                         const float* vW, const float* vb,
                                         const float* xyz, float* scr,
                                         int lane, v8f out[4]) {
  v8f sc = {0.f, 0.f, 0.f, 0.f, 0.f, 0.f, 0.f, 0.f};
#pragma unroll
  for (int h = 0; h < 2; ++h)
    sc = wmma_bf16(make_qA(qW, qb, xyz, h, lane), make_kB(kW, kb, xyz, h, lane), sc);

  // row-wise softmax over j (16 lanes of each half-wave), scale 1/sqrt(64)
  float prob[8];
#pragma unroll
  for (int r = 0; r < 8; ++r) {
    float x = sc[r] * 0.125f;
    float m = x;
#pragma unroll
    for (int msk = 1; msk < 16; msk <<= 1) m = fmaxf(m, __shfl_xor(m, msk, 32));
    float ex = __expf(x - m);
    float s = ex;
#pragma unroll
    for (int msk = 1; msk < 16; msk <<= 1) s += __shfl_xor(s, msk, 32);
    prob[r] = ex / s;
  }
  // transpose D layout -> A layout via LDS bounce
  int lg = lane >> 4, n = lane & 15;
#pragma unroll
  for (int r = 0; r < 8; ++r) scr[(r + 8 * lg) * 16 + n] = prob[r];
  lds_wait();
  union { v16bf v; __bf16 u[16]; } a;
  int jb = 8 * lg;
#pragma unroll
  for (int e = 0; e < 8; ++e) {
    a.u[e] = (__bf16)scr[n * 16 + jb + e];
    a.u[8 + e] = (__bf16)0.f;   // K padding 16..31
  }
#pragma unroll
  for (int t = 0; t < 4; ++t) {
    v8f acc = {0.f, 0.f, 0.f, 0.f, 0.f, 0.f, 0.f, 0.f};
    acc = wmma_bf16(a.v, make_vB(vW, vb, xyz, t, lane), acc);
#pragma unroll
    for (int r = 0; r < 8; ++r) acc[r] = fmaxf(acc[r], 0.f);
    out[t] = acc;
  }
}

__global__ __launch_bounds__(256) void pt_cross_attn(
    const float* __restrict__ g_xyz, const float* __restrict__ g_feat,
    const float* qW, const float* qb, const float* kW, const float* kb,
    const float* vW, const float* vb,
    const float* pmqW, const float* pmqb, const float* pmkW, const float* pmkb,
    const float* pmvW, const float* pmvb,
    const float* pbqW, const float* pbqb, const float* pbkW, const float* pbkb,
    const float* pbvW, const float* pbvb,
    const float* we1W, const float* we1b, const float* we2W, const float* we2b,
    const float* reW, const float* reb,
    float* __restrict__ out) {
  __shared__ __align__(16) float s_featf[CC * KN][MP]; // 68 KB f32, [c*16+k][m padded]
  __shared__ float s_xyz[MT][KN][4];                   //  4 KB
  __shared__ __bf16 s_w[4][CC * CC];                   // 32 KB: kW, vW, we1, we2 ([o][c])
  __shared__ __bf16 s_bounce[NW][KN * CC];             // 16 KB per-wave bf16 bounce
  __shared__ float s_f32[NW][KN * KN];                 //  8 KB per-wave f32 scratch
  __shared__ float s_reW[CC][CC];                      // 16 KB, transposed [c][o]

  const int tid = threadIdx.x;
  const int b   = blockIdx.x / (MM / MT);
  const int m0  = (blockIdx.x % (MM / MT)) * MT;

  // ---- async DMA of the feature tile straight into LDS (raw f32, native order) ----
  const float* fbase = g_feat + (size_t)b * CC * KN * MM + m0;
  {
    unsigned lbase = lds_offset(&s_featf[0][0]);
    // 1024 rows of 16 floats; each lane moves 16B (b128) chunks: 4096 chunks total
    for (int i = tid; i < CC * KN * 4; i += 256) {
      int row = i >> 2, jj = i & 3;                         // row = c*16+k, jj = m-quad
      unsigned goff = (unsigned)((row * MM + 4 * jj) * 4);  // bytes from fbase
      unsigned loff = lbase + (unsigned)((row * MP + 4 * jj) * 4);
      asm volatile("global_load_async_to_lds_b128 %0, %1, %2 offset:0"
                   :: "v"(loff), "v"(goff), "s"(fbase) : "memory");
    }
  }

  // ---- synchronous staging of the small stuff ----
  const float* xbase = g_xyz + (size_t)b * 3 * KN * MM + m0;
  for (int i = tid; i < 3 * KN * MT; i += 256) {
    int p = i & (MT - 1), k = (i >> 4) & 15, ch = i >> 8;
    s_xyz[p][k][ch] = xbase[(ch * KN + k) * MM + p];
  }
  for (int i = tid; i < CC * CC; i += 256) s_w[0][i] = (__bf16)kW[i];
  for (int i = tid; i < CC * CC; i += 256) s_w[1][i] = (__bf16)vW[i];
  for (int i = tid; i < CC * CC; i += 256) s_w[2][i] = (__bf16)we1W[i];
  for (int i = tid; i < CC * CC; i += 256) s_w[3][i] = (__bf16)we2W[i];
  for (int i = tid; i < CC * CC; i += 256) s_reW[i & 63][i >> 6] = reW[i];

  async_wait();       // our wave's async chunks have landed in LDS
  __syncthreads();    // everyone's have

  const int lane = tid & 31, wid = tid >> 5;
  const int lg = lane >> 4, n = lane & 15;
  __bf16* sbw = s_bounce[wid];
  float* scrw = s_f32[wid];

  for (int pp = 0; pp < PW; ++pp) {
    const int pt = wid * PW + pp;
    const float* xyz = &s_xyz[pt][0][0];

    v16bf fa0 = load_A_feat(&s_featf[0][0], pt, 0, lane);
    v16bf fa1 = load_A_feat(&s_featf[0][0], pt, 1, lane);

    // key = feat @ kW^T + kb ; value = feat @ vW^T + vb   (D layout)
    v8f keyD[4], valD[4];
#pragma unroll
    for (int t = 0; t < 4; ++t) {
      v8f ak = {0.f, 0.f, 0.f, 0.f, 0.f, 0.f, 0.f, 0.f};
      ak = wmma_bf16(fa0, load_Bw(s_w[0], t, 0, lane), ak);
      ak = wmma_bf16(fa1, load_Bw(s_w[0], t, 1, lane), ak);
      float bk = kb[16 * t + n];
      v8f av = {0.f, 0.f, 0.f, 0.f, 0.f, 0.f, 0.f, 0.f};
      av = wmma_bf16(fa0, load_Bw(s_w[1], t, 0, lane), av);
      av = wmma_bf16(fa1, load_Bw(s_w[1], t, 1, lane), av);
      float bv = vb[16 * t + n];
#pragma unroll
      for (int r = 0; r < 8; ++r) { ak[r] += bk; av[r] += bv; }
      keyD[t] = ak;
      valD[t] = av;
    }

    // rel = key - query   (query from 3-channel xyz, per-lane in D layout)
    v8f rel[4];
#pragma unroll
    for (int t = 0; t < 4; ++t) {
      int c = 16 * t + n;
      float w0 = qW[3 * c], w1 = qW[3 * c + 1], w2 = qW[3 * c + 2], b0 = qb[c];
#pragma unroll
      for (int r = 0; r < 8; ++r) {
        int row = r + 8 * lg;
        float q = b0 + xyz[row * 4] * w0 + xyz[row * 4 + 1] * w1 + xyz[row * 4 + 2] * w2;
        rel[t][r] = keyD[t][r] - q;
      }
    }

    v8f pem[4];
    self_att(pmqW, pmqb, pmkW, pmkb, pmvW, pmvb, xyz, scrw, lane, pem);
#pragma unroll
    for (int t = 0; t < 4; ++t)
#pragma unroll
      for (int r = 0; r < 8; ++r) rel[t][r] *= pem[t][r];

    v8f peb[4];
    self_att(pbqW, pbqb, pbkW, pbkb, pbvW, pbvb, xyz, scrw, lane, peb);
#pragma unroll
    for (int t = 0; t < 4; ++t)
#pragma unroll
      for (int r = 0; r < 8; ++r) {
        rel[t][r]  += peb[t][r];
        valD[t][r] += peb[t][r];
      }

    // weight_encoding: relu(rel @ we1^T + b1) @ we2^T + b2
    lds_wait();
#pragma unroll
    for (int t = 0; t < 4; ++t)
#pragma unroll
      for (int r = 0; r < 8; ++r)
        sbw[(r + 8 * lg) * CC + 16 * t + n] = (__bf16)rel[t][r];
    lds_wait();
    v16bf ra0 = load_A(sbw, 0, lane), ra1 = load_A(sbw, 1, lane);
    v8f w1o[4];
#pragma unroll
    for (int t = 0; t < 4; ++t) {
      v8f acc = {0.f, 0.f, 0.f, 0.f, 0.f, 0.f, 0.f, 0.f};
      acc = wmma_bf16(ra0, load_Bw(s_w[2], t, 0, lane), acc);
      acc = wmma_bf16(ra1, load_Bw(s_w[2], t, 1, lane), acc);
      float bb = we1b[16 * t + n];
#pragma unroll
      for (int r = 0; r < 8; ++r) acc[r] = fmaxf(acc[r] + bb, 0.f);
      w1o[t] = acc;
    }
#pragma unroll
    for (int t = 0; t < 4; ++t)
#pragma unroll
      for (int r = 0; r < 8; ++r)
        sbw[(r + 8 * lg) * CC + 16 * t + n] = (__bf16)w1o[t][r];
    lds_wait();
    ra0 = load_A(sbw, 0, lane);
    ra1 = load_A(sbw, 1, lane);
    v8f logit[4];
#pragma unroll
    for (int t = 0; t < 4; ++t) {
      v8f acc = {0.f, 0.f, 0.f, 0.f, 0.f, 0.f, 0.f, 0.f};
      acc = wmma_bf16(ra0, load_Bw(s_w[3], t, 0, lane), acc);
      acc = wmma_bf16(ra1, load_Bw(s_w[3], t, 1, lane), acc);
      float bb = we2b[16 * t + n];
#pragma unroll
      for (int r = 0; r < 8; ++r) acc[r] += bb;
      logit[t] = acc;
    }

    // softmax over K (8 VGPR rows + lane^16 partner), weighted sum with value, relu
    float featc[4];
#pragma unroll
    for (int t = 0; t < 4; ++t) {
      float mx = logit[t][0];
#pragma unroll
      for (int r = 1; r < 8; ++r) mx = fmaxf(mx, logit[t][r]);
      mx = fmaxf(mx, __shfl_xor(mx, 16, 32));
      float ex[8], sum = 0.f;
#pragma unroll
      for (int r = 0; r < 8; ++r) { ex[r] = __expf(logit[t][r] - mx); sum += ex[r]; }
      sum += __shfl_xor(sum, 16, 32);
      float inv = 1.f / sum;
      float acc = 0.f;
#pragma unroll
      for (int r = 0; r < 8; ++r) acc += ex[r] * inv * valD[t][r];
      acc += __shfl_xor(acc, 16, 32);
      featc[t] = fmaxf(acc, 0.f);
    }

    // residual: out = relu(feature) @ reW^T + reb  (VALU GEMV, reW staged transposed)
#pragma unroll
    for (int t = 0; t < 4; ++t) scrw[16 * t + n] = featc[t];  // dup writes, same value
    lds_wait();
    float a0 = reb[lane], a1 = reb[lane + 32];
#pragma unroll
    for (int c = 0; c < CC; ++c) {
      float f = scrw[c];
      a0 += f * s_reW[c][lane];
      a1 += f * s_reW[c][lane + 32];
    }
    size_t ob = ((size_t)b * MM + m0 + pt) * CC;
    out[ob + lane]      = a0;
    out[ob + lane + 32] = a1;
  }
}

extern "C" void kernel_launch(void* const* d_in, const int* in_sizes, int n_in,
                              void* d_out, int out_size, void* d_ws, size_t ws_size,
                              hipStream_t stream) {
  (void)in_sizes; (void)n_in; (void)out_size; (void)d_ws; (void)ws_size;
  const float* g_xyz  = (const float*)d_in[0];
  const float* g_feat = (const float*)d_in[1];
  const float* qW  = (const float*)d_in[2];  const float* qb  = (const float*)d_in[3];
  const float* kW  = (const float*)d_in[4];  const float* kb  = (const float*)d_in[5];
  const float* vW  = (const float*)d_in[6];  const float* vb  = (const float*)d_in[7];
  const float* pmqW = (const float*)d_in[8];  const float* pmqb = (const float*)d_in[9];
  const float* pmkW = (const float*)d_in[10]; const float* pmkb = (const float*)d_in[11];
  const float* pmvW = (const float*)d_in[12]; const float* pmvb = (const float*)d_in[13];
  const float* pbqW = (const float*)d_in[14]; const float* pbqb = (const float*)d_in[15];
  const float* pbkW = (const float*)d_in[16]; const float* pbkb = (const float*)d_in[17];
  const float* pbvW = (const float*)d_in[18]; const float* pbvb = (const float*)d_in[19];
  const float* we1W = (const float*)d_in[20]; const float* we1b = (const float*)d_in[21];
  const float* we2W = (const float*)d_in[22]; const float* we2b = (const float*)d_in[23];
  const float* reW  = (const float*)d_in[24]; const float* reb  = (const float*)d_in[25];
  float* out = (float*)d_out;

  dim3 grid((Bc * MM) / MT), block(256);
  hipLaunchKernelGGL(pt_cross_attn, grid, block, 0, stream,
                     g_xyz, g_feat, qW, qb, kW, kb, vW, vb,
                     pmqW, pmqb, pmkW, pmkb, pmvW, pmvb,
                     pbqW, pbqb, pbkW, pbkb, pbvW, pbvb,
                     we1W, we1b, we2W, we2b, reW, reb, out);
}
